// DeepHitLoss_3212635537826
// MI455X (gfx1250) — compile-verified, matrix operand order checked by reading, not run on hardware
//
#include <hip/hip_runtime.h>

typedef __attribute__((ext_vector_type(16))) _Float16 v16h;
typedef __attribute__((ext_vector_type(8)))  float    v8f;

#define ALPHA_C   0.5f
#define INV_SIGMA 10.0f     // 1 / 0.1
#define EPS_C     1e-7f
#define SBLK      256       // rows per sorted block

// ---------------------------------------------------------------------------
// K1: per-row scan. One wave (32 lanes) per row of 512.
// Computes cdf via wave scan, writes E = exp(cdf/sigma), bin_idx, cdf_at, nll.
// ---------------------------------------------------------------------------
__global__ void k_row_scan(const float* __restrict__ pmf,
                           const float* __restrict__ times,
                           const int*   __restrict__ events,
                           const float* __restrict__ time_bins,
                           float* __restrict__ E,
                           int*   __restrict__ bin_out,
                           float* __restrict__ cdfat_out,
                           float* __restrict__ nll_out,
                           int N, int T) {
  int lane = threadIdx.x & 31;
  int row  = blockIdx.x * (blockDim.x >> 5) + (threadIdx.x >> 5);
  if (row >= N) return;

  float t = times[row];
  // searchsorted(time_bins, t, 'left'): first idx with time_bins[idx] >= t
  int lo = 0, hi = T;
  while (lo < hi) { int mid = (lo + hi) >> 1; if (time_bins[mid] < t) lo = mid + 1; else hi = mid; }
  int b = lo - 1; if (b < 0) b = 0; if (b > T - 1) b = T - 1;

  // lane covers 16 contiguous columns
  const float* rp = pmf + (size_t)row * T + lane * 16;
  float v[16];
#pragma unroll
  for (int c = 0; c < 4; ++c) {
    float4 q = ((const float4*)rp)[c];
    v[4*c+0] = q.x; v[4*c+1] = q.y; v[4*c+2] = q.z; v[4*c+3] = q.w;
  }
  float lsum = 0.f;
#pragma unroll
  for (int e = 0; e < 16; ++e) lsum += v[e];

  // inclusive wave scan (wave32)
  float incl = lsum;
#pragma unroll
  for (int off = 1; off < 32; off <<= 1) {
    float nb = __shfl_up(incl, off);
    if (lane >= off) incl += nb;
  }
  float base  = incl - lsum;            // exclusive prefix
  float total = __shfl(incl, 31);       // full row sum (cdf[:, -1])

  float run = base, pa = 0.f, ca = 0.f;
  float* Erow = E + (size_t)row * T + lane * 16;
#pragma unroll
  for (int e = 0; e < 16; ++e) {
    run += v[e];                         // cdf value at column lane*16+e
    Erow[e] = __expf(run * INV_SIGMA);   // <= e^10 ~ 22026, safe in f32/f16
    if (lane * 16 + e == b) { pa = v[e]; ca = run; }
  }
  int owner = b >> 4;
  pa = __shfl(pa, owner);
  ca = __shfl(ca, owner);

  if (lane == 0) {
    int   ev   = events[row];
    float surv = total - ca + pa;
    float nll  = (ev == 1) ? -__logf(pa + EPS_C) : -__logf(surv + EPS_C);
    bin_out[row]   = b;
    cdfat_out[row] = ca;
    nll_out[row]   = nll;
  }
}

// ---------------------------------------------------------------------------
// K2: cnt[i] = #{j : t_j > t_i}; stable descending rank for the sort scatter.
// Tile of `times` is staged into LDS via the CDNA5 async global->LDS path
// (no VGPR round-trip), synchronized with s_wait_asynccnt + barrier.
// ---------------------------------------------------------------------------
__global__ void k_rank(const float* __restrict__ times,
                       int* __restrict__ cnt_out,
                       int* __restrict__ inv_out,
                       int N) {
  __shared__ float ts[256];
  int i = blockIdx.x * 256 + threadIdx.x;
  float ti = times[i];

  // Low 32 bits of a flat pointer into the LDS aperture are the LDS byte
  // address (CDNA5 aperture mapping).
  unsigned lds_addr = (unsigned)(size_t)&ts[threadIdx.x];

  int cnt = 0, tie_before = 0;
  for (int tile = 0; tile < N; tile += 256) {
    if (tile + 256 < N)
      __builtin_prefetch(&times[tile + 256 + threadIdx.x], 0, 3); // global_prefetch_b8

    unsigned long long ga =
        (unsigned long long)(size_t)&times[tile + threadIdx.x];
    // async copy: global -> LDS, tracked with ASYNCcnt
    asm volatile("global_load_async_to_lds_b32 %0, %1, off"
                 :: "v"(lds_addr), "v"(ga) : "memory");
    asm volatile("s_wait_asynccnt 0" ::: "memory");
    __syncthreads();

#pragma unroll 8
    for (int j = 0; j < 256; ++j) {
      float tj = ts[j];
      cnt        += (tj > ti) ? 1 : 0;
      tie_before += ((tj == ti) && (tile + j < i)) ? 1 : 0;
    }
    __syncthreads();
  }
  cnt_out[i] = cnt;
  inv_out[cnt + tie_before] = i;   // bijective rank -> permutation
}

// ---------------------------------------------------------------------------
// K3: ColSum[sb][k] = sum over the 256 rows of sorted block sb of E[row][k].
// Recast as ones(16x32) x E_block(32x16) on v_wmma_f32_16x16x32_f16.
// blockDim = 32 (one full wave, EXEC all-ones as WMMA requires).
// Fully unrolled: 8 independent back-to-back WMMAs, gathers overlap XDL work.
// ---------------------------------------------------------------------------
__global__ void k_colsum_wmma(const float* __restrict__ E,
                              const int*   __restrict__ inv,
                              float* __restrict__ ColSum,
                              int T) {
  __shared__ int sj[SBLK];
  int lane = threadIdx.x;       // 0..31
  int sb   = blockIdx.x;        // sorted block
  int ct   = blockIdx.y;        // 16-column tile
  for (int r = lane; r < SBLK; r += 32) sj[r] = inv[sb * SBLK + r];
  __syncthreads();

  int col = ct * 16 + (lane & 15);
  int grp = lane >> 4;          // lane group covers 16 of the 32 K rows

  v16h a;
#pragma unroll
  for (int k = 0; k < 16; ++k) a[k] = (_Float16)1.0f;   // ones A-matrix

  v8f acc = {};
#pragma unroll
  for (int chunk = 0; chunk < SBLK; chunk += 32) {
    v16h bmat;
#pragma unroll
    for (int k2 = 0; k2 < 16; ++k2) {
      int r = chunk + grp * 16 + k2;                    // bijection over 32 rows
      bmat[k2] = (_Float16)E[(size_t)sj[r] * T + col];
    }
    // D = 1 * B + C  -> every row of D = column sums of B
    acc = __builtin_amdgcn_wmma_f32_16x16x32_f16(
        false, a, false, bmat, (short)0, acc, false, false);
  }
  // C/D layout: VGPR0, lanes 0-15 hold D[0][lane]
  if (lane < 16) ColSum[(size_t)sb * T + col] = acc[0];
}

// ---------------------------------------------------------------------------
// K4: exclusive prefix over sorted blocks, per column.
// ---------------------------------------------------------------------------
__global__ void k_prefix(const float* __restrict__ ColSum,
                         float* __restrict__ Prefix,
                         int T, int NSB) {
  int k = blockIdx.x * blockDim.x + threadIdx.x;
  if (k >= T) return;
  float run = 0.f;
  for (int b = 0; b < NSB; ++b) {
    Prefix[(size_t)b * T + k] = run;
    run += ColSum[(size_t)b * T + k];
  }
}

// ---------------------------------------------------------------------------
// K5: within-block strict-comparison term + assemble per_i.
// ---------------------------------------------------------------------------
__global__ void k_pairwise(const float* __restrict__ E,
                           const float* __restrict__ times,
                           const int*   __restrict__ events,
                           const int*   __restrict__ inv,
                           const int*   __restrict__ bin,
                           const float* __restrict__ cdfat,
                           const int*   __restrict__ cnt,
                           const float* __restrict__ Prefix,
                           float* __restrict__ per_out,
                           int T) {
  __shared__ int   sj[SBLK];
  __shared__ float st[SBLK];
  int sb = blockIdx.x;
  int r  = threadIdx.x;
  int i  = inv[sb * SBLK + r];
  sj[r] = i;
  st[r] = times[i];
  __syncthreads();

  float ti = st[r];
  int   bi = bin[i];
  float s = 0.f;
  for (int rp = 0; rp < SBLK; ++rp) {
    float val = E[(size_t)sj[rp] * T + bi];
    s += (st[rp] > ti) ? val : 0.f;     // exact strict comparison in-block
  }
  s += Prefix[(size_t)sb * T + bi];     // earlier sorted blocks: strictly later times

  int   c    = cnt[i];
  float per  = __expf(-cdfat[i] * INV_SIGMA) * s / (float)(c > 0 ? c : 1);
  bool valid = (events[i] == 1) && (c > 0);
  per_out[i] = valid ? per : 0.f;
}

// ---------------------------------------------------------------------------
// K6: deterministic final reduction -> scalar loss.
// ---------------------------------------------------------------------------
__global__ void k_final(const float* __restrict__ nll,
                        const float* __restrict__ per,
                        const int*   __restrict__ events,
                        const int*   __restrict__ cnt,
                        float* __restrict__ out, int N) {
  __shared__ float s_nll[256], s_per[256];
  __shared__ int   s_ev[256],  s_vp[256];
  int tid = threadIdx.x;
  float an = 0.f, ap = 0.f; int ev = 0, vp = 0;
  for (int i = tid; i < N; i += 256) {
    an += nll[i];
    ap += per[i];
    int e = events[i];
    ev += e;
    vp += (e == 1 && cnt[i] > 0) ? 1 : 0;
  }
  s_nll[tid] = an; s_per[tid] = ap; s_ev[tid] = ev; s_vp[tid] = vp;
  __syncthreads();
  for (int off = 128; off > 0; off >>= 1) {
    if (tid < off) {
      s_nll[tid] += s_nll[tid + off];
      s_per[tid] += s_per[tid + off];
      s_ev[tid]  += s_ev[tid + off];
      s_vp[tid]  += s_vp[tid + off];
    }
    __syncthreads();
  }
  if (tid == 0) {
    float loss = s_nll[0] / (float)N;
    int np = s_vp[0];
    if ((s_ev[0] > 1) && (np > 0))      // ALPHA > 0 is compile-time true
      loss += ALPHA_C * s_per[0] / (float)np;
    out[0] = loss;
  }
}

// ---------------------------------------------------------------------------
extern "C" void kernel_launch(void* const* d_in, const int* in_sizes, int n_in,
                              void* d_out, int out_size, void* d_ws, size_t ws_size,
                              hipStream_t stream) {
  (void)n_in; (void)out_size; (void)ws_size;
  const float* pmf    = (const float*)d_in[0];
  const float* times  = (const float*)d_in[1];
  const int*   events = (const int*)  d_in[2];
  const float* tb     = (const float*)d_in[3];
  int N   = in_sizes[1];     // 8192
  int T   = in_sizes[3];     // 512
  int NSB = N / SBLK;        // 32 sorted blocks

  char* ws = (char*)d_ws;
  float* E      = (float*)ws; ws += (size_t)N * T * sizeof(float);   // 16 MB
  int*   bin    = (int*)  ws; ws += (size_t)N * sizeof(int);
  float* cdfat  = (float*)ws; ws += (size_t)N * sizeof(float);
  float* nll    = (float*)ws; ws += (size_t)N * sizeof(float);
  int*   cnt    = (int*)  ws; ws += (size_t)N * sizeof(int);
  int*   inv    = (int*)  ws; ws += (size_t)N * sizeof(int);
  float* per    = (float*)ws; ws += (size_t)N * sizeof(float);
  float* ColSum = (float*)ws; ws += (size_t)NSB * T * sizeof(float);
  float* Prefix = (float*)ws; ws += (size_t)NSB * T * sizeof(float);

  k_row_scan<<<N / 8, 256, 0, stream>>>(pmf, times, events, tb, E, bin, cdfat, nll, N, T);
  k_rank<<<N / 256, 256, 0, stream>>>(times, cnt, inv, N);
  dim3 g3(NSB, T / 16);
  k_colsum_wmma<<<g3, 32, 0, stream>>>(E, inv, ColSum, T);
  k_prefix<<<(T + 255) / 256, 256, 0, stream>>>(ColSum, Prefix, T, NSB);
  k_pairwise<<<NSB, SBLK, 0, stream>>>(E, times, events, inv, bin, cdfat, cnt, Prefix, per, T);
  k_final<<<1, 256, 0, stream>>>(nll, per, events, cnt, (float*)d_out, N);
}
